// MultiLevel_BlockAttention_42949673152
// MI455X (gfx1250) — matrix operand order repaired and checked by reading, not compiled, not run on hardware
//
#include <hip/hip_runtime.h>
#include <hip/hip_bf16.h>

// ---------- types ----------
typedef __attribute__((ext_vector_type(16))) __bf16 v16bf;
typedef __attribute__((ext_vector_type(8)))  float  v8f;

#define B_  4
#define S_  2048
#define E_  512
#define H_  8
#define DH_ 64
#define M_  (B_ * S_)              // 8192 rows of X
#define NX_ (B_ * S_ * E_)         // 4194304 elements of x / of out
#define NW_ (E_ * E_)              // 262144 per weight matrix
#define NQKV_ (B_ * H_ * S_ * DH_) // 4194304 per Q/K/V matrix

// ---------- helpers ----------
// native f32 -> bf16 (RNE fptrunc -> v_cvt_*bf16* on gfx1250)
__device__ __forceinline__ unsigned short bf16bits(float f) {
  return __builtin_bit_cast(unsigned short, static_cast<__bf16>(f));
}
__device__ __forceinline__ __bf16 bc(unsigned short h) {
  return __builtin_bit_cast(__bf16, h);
}

// A-matrix (16x32 bf16, MxK) fragment: lane holds M = lane&15.
// j<8  : K = kbase + off8 + j          (off8 = 8 for lanes 16..31)
// j>=8 : K = kbase + 16 + off8 + (j-8)
__device__ __forceinline__ v16bf load_afrag(const unsigned short* row, int kbase, int lane) {
  const int off = kbase + ((lane & 16) ? 8 : 0);
  v16bf v;
#pragma unroll
  for (int j = 0; j < 8; ++j) v[j] = bc(row[off + j]);
#pragma unroll
  for (int j = 0; j < 8; ++j) v[8 + j] = bc(row[off + 16 + j]);
  return v;
}

// B-matrix (32x16 bf16, KxN) fragment: lane holds N = lane&15.
// j<8: K = kbase + off8 + j ; j>=8: K = kbase + 16 + off8 + (j-8) (off8 = 8 for lanes 16..31); `row` = contiguous K-run for this N.
__device__ __forceinline__ v16bf load_bfrag(const unsigned short* row, int kbase, int lane) {
  const int off = kbase + ((lane & 16) ? 8 : 0);
  v16bf v;
#pragma unroll
  for (int j = 0; j < 8; ++j) v[j] = bc(row[off + j]);
#pragma unroll
  for (int j = 0; j < 8; ++j) v[8 + j] = bc(row[off + 16 + j]);
  return v;
}

__device__ __forceinline__ v8f wmma_bf16(v16bf a, v16bf b, v8f c) {
  return __builtin_amdgcn_wmma_f32_16x16x32_bf16(false, a, false, b, (short)0, c, false, false);
}

__device__ __forceinline__ float redMax16(float v) {
#pragma unroll
  for (int m = 1; m < 16; m <<= 1) v = fmaxf(v, __shfl_xor(v, m, 32));
  return v;
}
__device__ __forceinline__ float redSum16(float v) {
#pragma unroll
  for (int m = 1; m < 16; m <<= 1) v += __shfl_xor(v, m, 32);
  return v;
}

// ---------- kernel 1: f32 -> bf16 convert (packed stores) ----------
__global__ __launch_bounds__(256) void cvt_f32_bf16(const float* __restrict__ src,
                                                    unsigned short* __restrict__ dst, int n) {
  int i = (blockIdx.x * blockDim.x + threadIdx.x) * 4;
  if (i + 3 < n) {
    float4 v = *(const float4*)(src + i);
    unsigned short h[4] = {bf16bits(v.x), bf16bits(v.y), bf16bits(v.z), bf16bits(v.w)};
    uint2 p;
    p.x = (unsigned)h[0] | ((unsigned)h[1] << 16);
    p.y = (unsigned)h[2] | ((unsigned)h[3] << 16);
    *(uint2*)(dst + i) = p;
  }
}

// ---------- kernel 2: copy x into second tuple slot of d_out ----------
__global__ __launch_bounds__(256) void copy_f32(const float* __restrict__ src,
                                                float* __restrict__ dst, int n) {
  int i = (blockIdx.x * blockDim.x + threadIdx.x) * 4;
  if (i + 3 < n) *(float4*)(dst + i) = *(const float4*)(src + i);
}

// ---------- kernel 3: QKV projection GEMM ----------
// C[m, e] = sum_k X[m, k] * W[k, e]; written bf16 head-split [B,H,S,dh].
// grid: (M/128, E/64, 3). block: 256 threads = 8 waves; wave w owns rows w*16..w*16+15.
__global__ __launch_bounds__(256) void qkv_gemm(const unsigned short* __restrict__ xb,
                                                const unsigned short* __restrict__ Wb,
                                                unsigned short* __restrict__ qkv) {
  const int z = blockIdx.z;
  const unsigned short* __restrict__ W = Wb + (size_t)z * NW_;
  unsigned short* __restrict__ dst = qkv + (size_t)z * NQKV_;

  const int rowBase = blockIdx.x * 128;
  const int colBase = blockIdx.y * 64;
  const int tid = threadIdx.x, wv = tid >> 5, lane = tid & 31;

  __shared__ __align__(16) unsigned short As[128][40];   // X tile: 128 rows x 32 K (16B-aligned rows)
  __shared__ __align__(16) unsigned short BsT[64][40];   // W tile transposed: 64 cols x 32 K

  v8f acc[4];
#pragma unroll
  for (int t = 0; t < 4; ++t)
#pragma unroll
    for (int r = 0; r < 8; ++r) acc[t][r] = 0.f;

  for (int ks = 0; ks < 16; ++ks) {        // K = 512 in steps of 32
    __syncthreads();
    // As: 128x32 halfs = 512 chunks of 8 halfs (16B); 2 chunks/thread, B128 both sides
#pragma unroll
    for (int i = 0; i < 2; ++i) {
      int ch = tid + i * 256;
      int r = ch >> 2, c8 = (ch & 3) * 8;
      uint4 d = *(const uint4*)&xb[(size_t)(rowBase + r) * E_ + ks * 32 + c8];
      *(uint4*)&As[r][c8] = d;
    }
    // BsT: 32x64 halfs; B128 global load along e, transpose-scatter into LDS
    {
      int ch = tid;                        // 256 chunks of 8 halfs
      int k = ch >> 3, e8 = (ch & 7) * 8;
      uint4 d = *(const uint4*)&W[(size_t)(ks * 32 + k) * E_ + colBase + e8];
      const unsigned short* p = (const unsigned short*)&d;
#pragma unroll
      for (int j = 0; j < 8; ++j) BsT[e8 + j][k] = p[j];
    }
    __syncthreads();

    v16bf a = load_afrag(&As[wv * 16 + (lane & 15)][0], 0, lane);
#pragma unroll
    for (int t = 0; t < 4; ++t) {
      v16bf b = load_bfrag(&BsT[t * 16 + (lane & 15)][0], 0, lane);
      acc[t] = wmma_bf16(a, b, acc[t]);
    }
  }

  // C layout: VGPR r -> M = r (lanes 0-15) / r+8 (lanes 16-31); N = lane&15.
#pragma unroll
  for (int t = 0; t < 4; ++t) {
    int e = colBase + t * 16 + (lane & 15);
    int h = e >> 6, d = e & 63;
#pragma unroll
    for (int r = 0; r < 8; ++r) {
      int m = rowBase + wv * 16 + r + ((lane & 16) ? 8 : 0);
      int b = m >> 11, s = m & (S_ - 1);
      dst[(((size_t)(b * H_ + h) * S_) + s) * DH_ + d] = bf16bits(acc[t][r]);
    }
  }
}

// ---------- kernel 4: fused flash attention ----------
// grid: (S/64, B*H). block: 128 threads = 4 waves, wave w owns query rows w*16..w*16+15.
__global__ __launch_bounds__(128) void attention(const unsigned short* __restrict__ qkv,
                                                 float* __restrict__ out) {
  const int qt = blockIdx.x;               // query tile (64 queries)
  const int bh = blockIdx.y;               // b*H + h
  const int b = bh >> 3, h = bh & 7;
  const size_t headOff = (size_t)bh * S_ * DH_;
  const unsigned short* __restrict__ Qg = qkv + headOff;
  const unsigned short* __restrict__ Kg = qkv + (size_t)NQKV_ + headOff;
  const unsigned short* __restrict__ Vg = qkv + 2 * (size_t)NQKV_ + headOff;

  const int tid = threadIdx.x, wv = tid >> 5, lane = tid & 31;

  __shared__ __align__(16) unsigned short Ks[64][72];    // K tile   [key][dh]   (B frag for Q*K^T)
  __shared__ __align__(16) unsigned short Vst[64][72];   // V tile^T [dh][key]   (B frag for P*V)
  __shared__ __align__(16) unsigned short Ps[64][72];    // Q staging, then P tile [qrow][key]

  // stage Q tile (B128 both sides), build A fragments once
#pragma unroll
  for (int i = 0; i < 4; ++i) {
    int ch = tid + i * 128;                // 512 chunks of 8 halfs
    int r = ch >> 3, c8 = (ch & 7) * 8;
    uint4 d = *(const uint4*)&Qg[(size_t)(qt * 64 + r) * DH_ + c8];
    *(uint4*)&Ps[r][c8] = d;
  }
  __syncthreads();
  v16bf aQ[2];
  aQ[0] = load_afrag(&Ps[wv * 16 + (lane & 15)][0], 0, lane);
  aQ[1] = load_afrag(&Ps[wv * 16 + (lane & 15)][0], 32, lane);

  float m[8], l[8];
  v8f o[4];
#pragma unroll
  for (int r = 0; r < 8; ++r) { m[r] = -3.0e38f; l[r] = 0.f; }
#pragma unroll
  for (int t = 0; t < 4; ++t)
#pragma unroll
    for (int r = 0; r < 8; ++r) o[t][r] = 0.f;

  for (int kt = 0; kt < S_ / 64; ++kt) {
    __syncthreads();                        // previous tile fully consumed

    // K tile: async global -> LDS (no VGPR round-trip, tracked by ASYNCcnt)
    {
      unsigned long long kbase = (unsigned long long)(Kg + (size_t)kt * 64 * DH_);
#pragma unroll
      for (int i = 0; i < 4; ++i) {
        int ch = tid + i * 128;             // 512 chunks of 16 bytes
        int r = ch >> 3, cB = (ch & 7) * 16;
        unsigned voff = (unsigned)(r * (DH_ * 2) + cB);
        unsigned lds = (unsigned)(uintptr_t)&Ks[r][0] + (unsigned)cB;
        asm volatile("global_load_async_to_lds_b128 %0, %1, %2"
                     :: "v"(lds), "v"(voff), "s"(kbase)
                     : "memory");
      }
    }
    // V tile: B128 global load, transpose-scatter into LDS
#pragma unroll
    for (int i = 0; i < 4; ++i) {
      int ch = tid + i * 128;
      int r = ch >> 3, c8 = (ch & 7) * 8;
      uint4 d = *(const uint4*)&Vg[(size_t)(kt * 64 + r) * DH_ + c8];
      const unsigned short* p = (const unsigned short*)&d;
#pragma unroll
      for (int j = 0; j < 8; ++j) Vst[c8 + j][r] = p[j];
    }
    if (kt + 1 < S_ / 64) {
      __builtin_prefetch(&Kg[(size_t)(kt + 1) * 64 * DH_], 0, 0);
      __builtin_prefetch(&Vg[(size_t)(kt + 1) * 64 * DH_], 0, 0);
    }
    asm volatile("s_wait_asynccnt 0" ::: "memory");
    __syncthreads();

    // S = Q * K^T : B[kdim, key] = K[key][kdim] -> contiguous run in Ks row.
    v8f sc[4];
#pragma unroll
    for (int t = 0; t < 4; ++t)
#pragma unroll
      for (int r = 0; r < 8; ++r) sc[t][r] = 0.f;
#pragma unroll
    for (int ks = 0; ks < 2; ++ks)
#pragma unroll
      for (int t = 0; t < 4; ++t) {
        v16bf bK = load_bfrag(&Ks[t * 16 + (lane & 15)][0], ks * 32, lane);
        sc[t] = wmma_bf16(aQ[ks], bK, sc[t]);
      }

    // online softmax: row r lives in VGPR r across the 16 lanes of this half-wave
#pragma unroll
    for (int r = 0; r < 8; ++r) {
      float rm = fmaxf(fmaxf(sc[0][r], sc[1][r]), fmaxf(sc[2][r], sc[3][r]));
      rm = redMax16(rm);
      float mn = fmaxf(m[r], rm);
      float alpha = __expf(m[r] - mn);
      float rs = 0.f;
#pragma unroll
      for (int t = 0; t < 4; ++t) {
        float p = __expf(sc[t][r] - mn);
        sc[t][r] = p;
        rs += p;
      }
      rs = redSum16(rs);
      l[r] = l[r] * alpha + rs;
      m[r] = mn;
#pragma unroll
      for (int t = 0; t < 4; ++t) o[t][r] *= alpha;
    }

    // C layout -> A layout via LDS (same-wave rows only)
    const int rowsel = wv * 16 + ((lane & 16) ? 8 : 0);
#pragma unroll
    for (int r = 0; r < 8; ++r)
#pragma unroll
      for (int t = 0; t < 4; ++t)
        Ps[rowsel + r][t * 16 + (lane & 15)] = bf16bits(sc[t][r]);
    __syncthreads();

    // O += P * V : B[key, dh] -> Vst[dh][key] contiguous run.
#pragma unroll
    for (int ks = 0; ks < 2; ++ks) {
      v16bf aP = load_afrag(&Ps[wv * 16 + (lane & 15)][0], ks * 32, lane);
#pragma unroll
      for (int t = 0; t < 4; ++t) {
        v16bf bV = load_bfrag(&Vst[t * 16 + (lane & 15)][0], ks * 32, lane);
        o[t] = wmma_bf16(aP, bV, o[t]);
      }
    }
  }

  // epilogue: O / l, merge heads: out[b, s, h*64 + d]
#pragma unroll
  for (int r = 0; r < 8; ++r) {
    float inv = 1.0f / l[r];
    int srow = qt * 64 + wv * 16 + r + ((lane & 16) ? 8 : 0);
#pragma unroll
    for (int t = 0; t < 4; ++t) {
      int e = h * 64 + t * 16 + (lane & 15);
      out[((size_t)(b * S_ + srow)) * E_ + e] = o[t][r] * inv;
    }
  }
}

// ---------- launcher ----------
extern "C" void kernel_launch(void* const* d_in, const int* in_sizes, int n_in,
                              void* d_out, int out_size, void* d_ws, size_t ws_size,
                              hipStream_t stream) {
  (void)in_sizes; (void)n_in; (void)out_size; (void)ws_size;
  const float* x  = (const float*)d_in[0];
  const float* Wq = (const float*)d_in[1];
  const float* Wk = (const float*)d_in[2];
  const float* Wv = (const float*)d_in[3];
  float* out = (float*)d_out;

  // workspace layout (bf16 stored as ushort)
  unsigned short* xb  = (unsigned short*)d_ws;          // NX_
  unsigned short* Wb  = xb + (size_t)NX_;               // 3 * NW_
  unsigned short* qkv = Wb + 3 * (size_t)NW_;           // 3 * NQKV_

  // 1) conversions + tuple second element (x passthrough)
  cvt_f32_bf16<<<NX_ / 4 / 256, 256, 0, stream>>>(x, xb, NX_);
  cvt_f32_bf16<<<NW_ / 4 / 256, 256, 0, stream>>>(Wq, Wb + 0 * (size_t)NW_, NW_);
  cvt_f32_bf16<<<NW_ / 4 / 256, 256, 0, stream>>>(Wk, Wb + 1 * (size_t)NW_, NW_);
  cvt_f32_bf16<<<NW_ / 4 / 256, 256, 0, stream>>>(Wv, Wb + 2 * (size_t)NW_, NW_);
  copy_f32<<<NX_ / 4 / 256, 256, 0, stream>>>(x, out + (size_t)NX_, NX_);

  // 2) Q/K/V projections
  qkv_gemm<<<dim3(M_ / 128, E_ / 64, 3), 256, 0, stream>>>(xb, Wb, qkv);

  // 3) fused attention
  attention<<<dim3(S_ / 64, B_ * H_), 128, 0, stream>>>(qkv, out);
}
